// Transformer_86612310491636
// MI455X (gfx1250) — matrix operand order repaired and checked by reading, not compiled, not run on hardware
//
#include <hip/hip_runtime.h>
#include <hip/hip_bf16.h>
#include <math.h>

// ---------------------------------------------------------------------------
// Types for CDNA5 WMMA (wave32): v_wmma_f32_16x16x32_bf16
// ---------------------------------------------------------------------------
typedef __bf16 bf16;
typedef __bf16 bf16x8  __attribute__((ext_vector_type(8)));
typedef __bf16 bf16x16 __attribute__((ext_vector_type(16)));
typedef float  f32x8   __attribute__((ext_vector_type(8)));

union FragU { bf16x16 v; bf16x8 h[2]; };

#define WMMA_BF16(a, b, c) \
  __builtin_amdgcn_wmma_f32_16x16x32_bf16(false, (a), false, (b), (short)0, (c), false, false)

// ---------------------------------------------------------------------------
// gfx1250 async global->LDS copy (GLOBAL_LOAD_ASYNC_TO_LDS_B128, ASYNCcnt),
// guarded so compilation never regresses if the builtin is absent.
// Builtin expects int4-typed pointers: global (AS1) src, LDS (AS3) dst.
// ---------------------------------------------------------------------------
#if defined(__has_builtin)
#if __has_builtin(__builtin_amdgcn_global_load_async_to_lds_b128) && \
    __has_builtin(__builtin_amdgcn_s_wait_asynccnt)
#define HAVE_ASYNC_LDS 1
#endif
#endif

#ifdef HAVE_ASYNC_LDS
typedef int v4i_vs __attribute__((vector_size(16)));
typedef __attribute__((address_space(1))) v4i_vs g_v4i;
typedef __attribute__((address_space(3))) v4i_vs l_v4i;
// 16-byte global->LDS async copy (per-lane addresses, like a normal VMEM op)
__device__ __forceinline__ void async_copy_b128(const void* g, void* l) {
  __builtin_amdgcn_global_load_async_to_lds_b128(
      (g_v4i*)(size_t)g, (l_v4i*)(unsigned int)(size_t)l, 0, 0);
}
__device__ __forceinline__ void async_wait0() {
  __builtin_amdgcn_s_wait_asynccnt(0);
}
#else
__device__ __forceinline__ void async_copy_b128(const void* g, void* l) {
  *(uint4*)l = *(const uint4*)g;
}
__device__ __forceinline__ void async_wait0() {}
#endif

__device__ __forceinline__ f32x8 zero8() {
  f32x8 z;
#pragma unroll
  for (int i = 0; i < 8; ++i) z[i] = 0.0f;
  return z;
}

// Problem constants (from reference)
#define NLAYERS 6
#define BB      2
#define SS      2048
#define DD      1024
#define DFF     4096
#define NH      16
#define DH      64
#define MM      (BB * SS)   // 4096 rows of activations

// ---------------------------------------------------------------------------
// Elementwise helpers
// ---------------------------------------------------------------------------
__global__ void cvt_f32_to_bf16_kernel(const float* __restrict__ src,
                                       bf16* __restrict__ dst, int n) {
  int i = blockIdx.x * blockDim.x + threadIdx.x;
  if (i < n) dst[i] = (bf16)src[i];
}

__global__ void copy_f32_kernel(const float* __restrict__ src,
                                float* __restrict__ dst, int n) {
  int i = blockIdx.x * blockDim.x + threadIdx.x;
  if (i < n) dst[i] = src[i];
}

// ---------------------------------------------------------------------------
// Tiled bf16 WMMA GEMM:  out[M,N] = A[M,K] * B[K,N] (+bias, +relu / +residual)
// Block: 256 threads (8 waves). Tile 128x128x32. Wave tile 64x32 (4x2 WMMA).
// A tile staged via async global->LDS; B tile staged transposed through VGPRs.
// MODE 0: bf16 out = acc + bias
// MODE 1: bf16 out = relu(acc + bias)
// MODE 2: f32  out = acc + bias + residual
// ---------------------------------------------------------------------------
template <int MODE>
__global__ __launch_bounds__(256) void gemm_bf16_wmma(
    const bf16* __restrict__ A, const bf16* __restrict__ Bw,
    const float* __restrict__ bias, const float* __restrict__ res,
    void* __restrict__ out, int Mdim, int Ndim, int Kdim) {
  __shared__ __align__(16) bf16 sA[128][32];
  __shared__ __align__(16) bf16 sB[128][32];  // stored transposed: sB[n][k]

  const int tid  = threadIdx.x;
  const int lane = tid & 31;
  const int wave = tid >> 5;   // 0..7
  const int wm   = wave >> 2;  // 0..1  (64-row strip)
  const int wn   = wave & 3;   // 0..3  (32-col strip)
  const int m0   = blockIdx.y * 128;
  const int n0   = blockIdx.x * 128;
  const int lr   = lane & 15;
  const int hk   = (lane >> 4) << 3;  // 0/8   : A-frag K interleave
  const int kb   = (lane >> 4) << 4;  // 0/16  : unused; B-frag also uses hk interleave
  const int rsel = (lane >> 4) << 3;  // 0/8   : C row select

  f32x8 acc[4][2];
#pragma unroll
  for (int mt = 0; mt < 4; ++mt)
#pragma unroll
    for (int nt = 0; nt < 2; ++nt) acc[mt][nt] = zero8();

  // cooperative load indices
  const int arow = tid >> 1, acol = (tid & 1) * 16;  // A: 128 rows, 2 thr/row
  const int brow = tid >> 3, bcol = (tid & 7) * 16;  // B: 32 rows, 8 thr/row

  for (int k0 = 0; k0 < Kdim; k0 += 32) {
    __syncthreads();  // protect LDS from previous iteration's readers
    {
      // A tile: async DMA straight into LDS (no VGPR round trip)
      const bf16* gA = A + (size_t)(m0 + arow) * Kdim + k0 + acol;
      async_copy_b128(gA,     &sA[arow][acol]);
      async_copy_b128(gA + 8, &sA[arow][acol + 8]);

      // B tile: load through VGPRs, store transposed
      const bf16* gB = Bw + (size_t)(k0 + brow) * Ndim + n0 + bcol;
      bf16x8 b0 = *(const bf16x8*)gB;
      bf16x8 b1 = *(const bf16x8*)(gB + 8);
#pragma unroll
      for (int j = 0; j < 8; ++j) {
        sB[bcol + j][brow]     = b0[j];
        sB[bcol + 8 + j][brow] = b1[j];
      }

      // prefetch next K-step tiles into cache hierarchy
      if (k0 + 32 < Kdim) {
        __builtin_prefetch(gA + 32, 0, 3);
        __builtin_prefetch(gB + (size_t)32 * Ndim, 0, 3);
      }
    }
    async_wait0();
    __syncthreads();

    bf16x16 af[4];
#pragma unroll
    for (int mt = 0; mt < 4; ++mt) {
      int m = wm * 64 + mt * 16 + lr;
      FragU u;
      u.h[0] = *(const bf16x8*)&sA[m][hk];
      u.h[1] = *(const bf16x8*)&sA[m][16 + hk];
      af[mt] = u.v;
    }
#pragma unroll
    for (int nt = 0; nt < 2; ++nt) {
      int n = wn * 32 + nt * 16 + lr;
      FragU u;
      u.h[0] = *(const bf16x8*)&sB[n][hk];
      u.h[1] = *(const bf16x8*)&sB[n][16 + hk];
      bf16x16 bfv = u.v;
#pragma unroll
      for (int mt = 0; mt < 4; ++mt)
        acc[mt][nt] = WMMA_BF16(af[mt], bfv, acc[mt][nt]);
    }
  }

  // epilogue
#pragma unroll
  for (int mt = 0; mt < 4; ++mt) {
#pragma unroll
    for (int nt = 0; nt < 2; ++nt) {
      const int gn = n0 + wn * 32 + nt * 16 + lr;
      const float bv = bias[gn];
#pragma unroll
      for (int r = 0; r < 8; ++r) {
        const int gm = m0 + wm * 64 + mt * 16 + r + rsel;
        float v = acc[mt][nt][r] + bv;
        const size_t idx = (size_t)gm * Ndim + gn;
        if (MODE == 1) v = v > 0.0f ? v : 0.0f;
        if (MODE == 2)
          ((float*)out)[idx] = v + res[idx];
        else
          ((bf16*)out)[idx] = (bf16)v;
      }
    }
  }
}

// ---------------------------------------------------------------------------
// Flash attention, one (64-query tile, head, batch) per block. 128 threads =
// 4 waves; each wave owns 16 query rows. Q,K,V,O layout: [B, S, NH*DH].
// K tile staged via async global->LDS; V tile transposed through VGPRs.
// ---------------------------------------------------------------------------
__global__ __launch_bounds__(128) void flash_attn_wmma(
    const bf16* __restrict__ Q, const bf16* __restrict__ K,
    const bf16* __restrict__ V, bf16* __restrict__ O) {
  __shared__ __align__(16) bf16 sK[64][64];
  __shared__ __align__(16) bf16 sVt[64][64];  // sVt[dh][key]
  __shared__ __align__(16) bf16 sP[64][64];

  const int tid  = threadIdx.x;
  const int lane = tid & 31;
  const int wave = tid >> 5;  // 0..3
  const int lr   = lane & 15;
  const int hk   = (lane >> 4) << 3;
  const int kb   = (lane >> 4) << 4;
  const int rsel = (lane >> 4) << 3;

  const int qt = blockIdx.x;  // query tile
  const int h  = blockIdx.y;
  const int b  = blockIdx.z;

  const size_t rs = (size_t)NH * DH;  // 1024 row stride
  const bf16* qb = Q + (size_t)b * SS * rs + (size_t)h * DH;
  const bf16* kp = K + (size_t)b * SS * rs + (size_t)h * DH;
  const bf16* vp = V + (size_t)b * SS * rs + (size_t)h * DH;
  bf16* ob       = O + (size_t)b * SS * rs + (size_t)h * DH;

  // Q fragments in registers: 16 rows x 64 (two 32-wide K-steps)
  bf16x16 qf[2];
#pragma unroll
  for (int ks = 0; ks < 2; ++ks) {
    const int m     = qt * 64 + wave * 16 + lr;
    const bf16* p   = qb + (size_t)m * rs + ks * 32;
    FragU u;
    u.h[0] = *(const bf16x8*)(p + hk);
    u.h[1] = *(const bf16x8*)(p + 16 + hk);
    qf[ks] = u.v;
  }

  f32x8 oa[4];
#pragma unroll
  for (int nt = 0; nt < 4; ++nt) oa[nt] = zero8();
  float mrow[8], lsum[8];
#pragma unroll
  for (int r = 0; r < 8; ++r) {
    mrow[r] = -INFINITY;
    lsum[r] = 0.0f;
  }
  const float scale = 0.03125f;  // 1/sqrt(1024), dk_total per reference

  for (int kt = 0; kt < SS / 64; ++kt) {
    __syncthreads();
    {  // stage K tile (async, as-is) and V tile (transposed) into LDS
      const int row = tid >> 1, c = (tid & 1) * 32;
      const bf16* gk = kp + (size_t)(kt * 64 + row) * rs + c;
#pragma unroll
      for (int j = 0; j < 4; ++j)
        async_copy_b128(gk + 8 * j, &sK[row][c + 8 * j]);

      const bf16* gv = vp + (size_t)(kt * 64 + row) * rs + c;
#pragma unroll
      for (int j = 0; j < 32; ++j) sVt[c + j][row] = gv[j];

      if (kt + 1 < SS / 64) {
        __builtin_prefetch(gk + 64 * rs, 0, 3);
        __builtin_prefetch(gv + 64 * rs, 0, 3);
      }
    }
    async_wait0();
    __syncthreads();

    // S = scale * Q Kt   (4 column tiles of 16 keys)
    f32x8 st[4];
#pragma unroll
    for (int nt = 0; nt < 4; ++nt) {
      f32x8 s = zero8();
#pragma unroll
      for (int ks = 0; ks < 2; ++ks) {
        const int n = nt * 16 + lr;
        FragU u;
        u.h[0] = *(const bf16x8*)&sK[n][ks * 32 + hk];
        u.h[1] = *(const bf16x8*)&sK[n][ks * 32 + 16 + hk];
        s = WMMA_BF16(qf[ks], u.v, s);
      }
#pragma unroll
      for (int r = 0; r < 8; ++r) s[r] *= scale;
      st[nt] = s;
    }

    // online softmax: each VGPR slot r is one row, spread over 16 lanes
#pragma unroll
    for (int r = 0; r < 8; ++r) {
      float mx = fmaxf(fmaxf(st[0][r], st[1][r]), fmaxf(st[2][r], st[3][r]));
#pragma unroll
      for (int off = 8; off >= 1; off >>= 1)
        mx = fmaxf(mx, __shfl_xor(mx, off, 32));
      const float mnew  = fmaxf(mrow[r], mx);
      const float alpha = __expf(mrow[r] - mnew);
      float rsum = 0.0f;
#pragma unroll
      for (int nt = 0; nt < 4; ++nt) {
        const float p = __expf(st[nt][r] - mnew);
        st[nt][r] = p;
        rsum += p;
      }
#pragma unroll
      for (int off = 8; off >= 1; off >>= 1)
        rsum += __shfl_xor(rsum, off, 32);
      lsum[r] = lsum[r] * alpha + rsum;
      mrow[r] = mnew;
#pragma unroll
      for (int nt = 0; nt < 4; ++nt) oa[nt][r] *= alpha;
    }

    // P -> LDS (bf16) to re-shape C-layout into A-fragment layout
#pragma unroll
    for (int nt = 0; nt < 4; ++nt)
#pragma unroll
      for (int r = 0; r < 8; ++r)
        sP[wave * 16 + r + rsel][nt * 16 + lr] = (bf16)st[nt][r];
    __syncthreads();

    // O += P V
#pragma unroll
    for (int nt = 0; nt < 4; ++nt) {
#pragma unroll
      for (int ks = 0; ks < 2; ++ks) {
        const int m = wave * 16 + lr;
        FragU ua;
        ua.h[0] = *(const bf16x8*)&sP[m][ks * 32 + hk];
        ua.h[1] = *(const bf16x8*)&sP[m][ks * 32 + 16 + hk];
        const int n = nt * 16 + lr;
        FragU ub;
        ub.h[0] = *(const bf16x8*)&sVt[n][ks * 32 + hk];
        ub.h[1] = *(const bf16x8*)&sVt[n][ks * 32 + 16 + hk];
        oa[nt] = WMMA_BF16(ua.v, ub.v, oa[nt]);
      }
    }
  }

  // normalize and store
#pragma unroll
  for (int nt = 0; nt < 4; ++nt)
#pragma unroll
    for (int r = 0; r < 8; ++r) {
      const int gm = qt * 64 + wave * 16 + r + rsel;
      const float v = oa[nt][r] / lsum[r];
      ob[(size_t)gm * rs + nt * 16 + lr] = (bf16)v;
    }
}

// ---------------------------------------------------------------------------
// LayerNorm: one block (256 threads) per row of D=1024
// ---------------------------------------------------------------------------
__global__ __launch_bounds__(256) void layernorm_kernel(
    const float* __restrict__ src, const float* __restrict__ g,
    const float* __restrict__ bta, float* __restrict__ dst) {
  __shared__ float red[256];
  const int row = blockIdx.x;
  const float* p = src + (size_t)row * DD;

  float s = 0.0f;
  for (int j = threadIdx.x; j < DD; j += 256) s += p[j];
  red[threadIdx.x] = s;
  __syncthreads();
  for (int off = 128; off >= 1; off >>= 1) {
    if (threadIdx.x < off) red[threadIdx.x] += red[threadIdx.x + off];
    __syncthreads();
  }
  const float mu = red[0] / DD;
  __syncthreads();

  float vs = 0.0f;
  for (int j = threadIdx.x; j < DD; j += 256) {
    const float d = p[j] - mu;
    vs += d * d;
  }
  red[threadIdx.x] = vs;
  __syncthreads();
  for (int off = 128; off >= 1; off >>= 1) {
    if (threadIdx.x < off) red[threadIdx.x] += red[threadIdx.x + off];
    __syncthreads();
  }
  const float rstd = rsqrtf(red[0] / DD + 1e-5f);

  for (int j = threadIdx.x; j < DD; j += 256)
    dst[(size_t)row * DD + j] = (p[j] - mu) * rstd * g[j] + bta[j];
}

// ---------------------------------------------------------------------------
// Host orchestration
// ---------------------------------------------------------------------------
static inline void cvt_launch(const float* s, bf16* d, size_t n, hipStream_t st) {
  const int nn = (int)n;
  cvt_f32_to_bf16_kernel<<<(nn + 255) / 256, 256, 0, st>>>(s, d, nn);
}

extern "C" void kernel_launch(void* const* d_in, const int* in_sizes, int n_in,
                              void* d_out, int out_size, void* d_ws,
                              size_t ws_size, hipStream_t stream) {
  const float* x    = (const float*)d_in[0];
  const float* Wq   = (const float*)d_in[1];
  const float* bq   = (const float*)d_in[2];
  const float* Wk   = (const float*)d_in[3];
  const float* bk   = (const float*)d_in[4];
  const float* Wv   = (const float*)d_in[5];
  const float* bv   = (const float*)d_in[6];
  const float* Wo   = (const float*)d_in[7];
  const float* bo   = (const float*)d_in[8];
  const float* ln1g = (const float*)d_in[9];
  const float* ln1b = (const float*)d_in[10];
  const float* W1   = (const float*)d_in[11];
  const float* b1   = (const float*)d_in[12];
  const float* W2   = (const float*)d_in[13];
  const float* b2   = (const float*)d_in[14];
  const float* ln2g = (const float*)d_in[15];
  const float* ln2b = (const float*)d_in[16];

  const size_t eW  = (size_t)NLAYERS * DD * DD;    // per projection tensor
  const size_t eFF = (size_t)NLAYERS * DD * DFF;   // per FFN tensor
  const size_t eX  = (size_t)MM * DD;
  const size_t eH  = (size_t)MM * DFF;

  char* ws = (char*)d_ws;
  size_t off = 0;
  auto alloc = [&](size_t bytes) -> void* {
    void* p = ws + off;
    off += (bytes + 255) & ~(size_t)255;
    return p;
  };
  bf16* wq_b = (bf16*)alloc(eW * 2);
  bf16* wk_b = (bf16*)alloc(eW * 2);
  bf16* wv_b = (bf16*)alloc(eW * 2);
  bf16* wo_b = (bf16*)alloc(eW * 2);
  bf16* w1_b = (bf16*)alloc(eFF * 2);
  bf16* w2_b = (bf16*)alloc(eFF * 2);
  float* x_cur = (float*)alloc(eX * 4);
  bf16* x_b    = (bf16*)alloc(eX * 2);
  bf16* q_b    = (bf16*)alloc(eX * 2);
  bf16* k_b    = (bf16*)alloc(eX * 2);
  bf16* v_b    = (bf16*)alloc(eX * 2);
  bf16* o_b    = (bf16*)alloc(eX * 2);
  bf16* h_b    = (bf16*)alloc(eH * 2);
  float* tmp_f = (float*)alloc(eH * 4);
  (void)ws_size; (void)in_sizes; (void)n_in;

  // weights -> bf16 (deterministic, every call)
  cvt_launch(Wq, wq_b, eW, stream);
  cvt_launch(Wk, wk_b, eW, stream);
  cvt_launch(Wv, wv_b, eW, stream);
  cvt_launch(Wo, wo_b, eW, stream);
  cvt_launch(W1, w1_b, eFF, stream);
  cvt_launch(W2, w2_b, eFF, stream);

  copy_f32_kernel<<<((int)eX + 255) / 256, 256, 0, stream>>>(x, x_cur, (int)eX);

  const dim3 gProj(DD / 128, MM / 128);    // N=1024
  const dim3 gFF1(DFF / 128, MM / 128);    // N=4096
  const dim3 gAttn(SS / 64, NH, BB);

  for (int i = 0; i < NLAYERS; ++i) {
    const bf16* wq = wq_b + (size_t)i * DD * DD;
    const bf16* wk = wk_b + (size_t)i * DD * DD;
    const bf16* wv = wv_b + (size_t)i * DD * DD;
    const bf16* wo = wo_b + (size_t)i * DD * DD;
    const bf16* w1 = w1_b + (size_t)i * DD * DFF;
    const bf16* w2 = w2_b + (size_t)i * DFF * DD;

    cvt_launch(x_cur, x_b, eX, stream);

    gemm_bf16_wmma<0><<<gProj, 256, 0, stream>>>(
        x_b, wq, bq + (size_t)i * DD, nullptr, q_b, MM, DD, DD);
    gemm_bf16_wmma<0><<<gProj, 256, 0, stream>>>(
        x_b, wk, bk + (size_t)i * DD, nullptr, k_b, MM, DD, DD);
    gemm_bf16_wmma<0><<<gProj, 256, 0, stream>>>(
        x_b, wv, bv + (size_t)i * DD, nullptr, v_b, MM, DD, DD);

    flash_attn_wmma<<<gAttn, 128, 0, stream>>>(q_b, k_b, v_b, o_b);

    gemm_bf16_wmma<2><<<gProj, 256, 0, stream>>>(
        o_b, wo, bo + (size_t)i * DD, x_cur, tmp_f, MM, DD, DD);
    layernorm_kernel<<<MM, 256, 0, stream>>>(
        tmp_f, ln1g + (size_t)i * DD, ln1b + (size_t)i * DD, x_cur);

    cvt_launch(x_cur, x_b, eX, stream);

    gemm_bf16_wmma<1><<<gFF1, 256, 0, stream>>>(
        x_b, w1, b1 + (size_t)i * DFF, nullptr, h_b, MM, DFF, DD);
    gemm_bf16_wmma<2><<<gProj, 256, 0, stream>>>(
        h_b, w2, b2 + (size_t)i * DD, x_cur, tmp_f, MM, DD, DFF);
    layernorm_kernel<<<MM, 256, 0, stream>>>(
        tmp_f, ln2g + (size_t)i * DD, ln2b + (size_t)i * DD, x_cur);
  }

  copy_f32_kernel<<<(out_size + 255) / 256, 256, 0, stream>>>(
      x_cur, (float*)d_out, out_size);
}
